// Net_45638322487525
// MI455X (gfx1250) — compile-verified, matrix operand order checked by reading, not run on hardware
//
#include <hip/hip_runtime.h>
#include <cstdint>
#include <cstddef>

// ---------------- problem constants ----------------
#define ANGN 9
#define AA   81
#define DDN  9
#define FEATC 8
#define AGGC 160
#define BB   2
#define HN   64
#define WN   64
#define HW   4096
#define NIMG (BB*AA)          // 162
#define M3D  (BB*DDN*HW)      // 73728
#define K3D  (AGGC*27)        // 4320
#define KSQ  (AA*FEATC)       // 648
#define KSQR 656              // padded psv row stride (16B-aligned chunks)
#define K2D  144
#define EPSF 1e-5f
#define BN_INV 0.99999500003749977f   // 1/sqrt(1+1e-5)

typedef __attribute__((ext_vector_type(16))) __bf16 v16bf;
typedef __attribute__((ext_vector_type(8)))  __bf16 v8bf;
typedef __attribute__((ext_vector_type(8)))  float  v8f;

union FragU { v16bf v; v8bf h[2]; };

__device__ __forceinline__ v8bf zero8() {
  v8bf z;
#pragma unroll
  for (int e = 0; e < 8; ++e) z[e] = (__bf16)0.0f;
  return z;
}

// ---------------- weight packing into WMMA fragment order ----------------
// w fp32 [Cout][Cin][KT]; K index k = t*Cin + ci.
// out[((nt*KB + kbi)*32 + lane)*16 + e] = w at kk = kbi*32 + koff(lane,e), co = nt*16 + (lane&15)
// koff(l,e) = ((e<8)? e : e+8) + ((l&16)? 8 : 0)   (CDNA5 16-bit B-fragment layout)
__global__ void pack_b_frag_kernel(const float* __restrict__ w, __bf16* __restrict__ out,
                                   int Cout, int Cin, int KT, int K, int KB, int n) {
  int i = blockIdx.x * 256 + threadIdx.x;
  if (i >= n) return;
  int e = i & 15;
  int l = (i >> 4) & 31;
  int rest = i >> 9;
  int kbi = rest % KB;
  int nt  = rest / KB;
  int koff = ((e < 8) ? e : (e + 8)) + ((l & 16) ? 8 : 0);
  int kk = kbi * 32 + koff;
  int co = nt * 16 + (l & 15);
  float v = 0.f;
  if (kk < K && co < Cout) {
    int t  = kk / Cin;
    int ci = kk - t * Cin;
    v = w[((long)co * Cin + ci) * KT + t];
  }
  out[i] = (__bf16)v;
}

// ---------------- mask (warp + photometric) ----------------
__global__ void mask_kernel(const float* __restrict__ lf, const float* __restrict__ disp,
                            float* __restrict__ mask) {
  int i = blockIdx.x * 256 + threadIdx.x;
  if (i >= BB * AA * HW) return;
  int p = i & 4095;
  int a = (i >> 12) % AA;
  int b = i / (AA * HW);
  int h = p >> 6, w = p & 63;
  float dsp = -disp[b * HW + p];
  float du = (float)(a / ANGN - 4);
  float dv = (float)(a % ANGN - 4);
  const float* img = lf + (long)(b * AA + a) * HW;
  float v;
  if (a == 40) {
    v = img[p];
  } else {
    float ix = (float)w * (64.f / 63.f) + dv * dsp - 0.5f;
    float iy = (float)h * (64.f / 63.f) + du * dsp - 0.5f;
    float x0f = floorf(ix), y0f = floorf(iy);
    int x0 = (int)x0f, y0 = (int)y0f;
    float wx = ix - x0f, wy = iy - y0f;
    float g00 = 0.f, g01 = 0.f, g10 = 0.f, g11 = 0.f;
    bool xa = (x0 >= 0) && (x0 < WN), xb2 = (x0 + 1 >= 0) && (x0 + 1 < WN);
    bool ya = (y0 >= 0) && (y0 < HN), yb2 = (y0 + 1 >= 0) && (y0 + 1 < HN);
    if (ya && xa)   g00 = img[y0 * WN + x0];
    if (ya && xb2)  g01 = img[y0 * WN + x0 + 1];
    if (yb2 && xa)  g10 = img[(y0 + 1) * WN + x0];
    if (yb2 && xb2) g11 = img[(y0 + 1) * WN + x0 + 1];
    v = g00 * (1.f - wy) * (1.f - wx) + g01 * (1.f - wy) * wx
      + g10 * wy * (1.f - wx) + g11 * wy * wx;
  }
  float ref = lf[(long)(b * AA + 40) * HW + p];
  float t = 1.f - fabsf(v - ref);
  mask[i] = t * t;
}

// ---------------- layer0 conv2d 1->16 (per-element A gather, tiny K) ----------------
// in: lf fp32 [img][64x64]; bp: fragment-packed bf16 (KB=1); out: bf16 NHWC [img][4096][16]
__global__ void conv2d_c1_kernel(const float* __restrict__ in, const __bf16* __restrict__ bp,
                                 const float* __restrict__ bng, const float* __restrict__ bnb,
                                 __bf16* __restrict__ out) {
  int lane  = threadIdx.x;
  int img   = blockIdx.y;
  int mbase = blockIdx.x * 16;
  int h = mbase >> 6, w0 = mbase & 63;
  int wl = w0 + (lane & 15);
  int off = (lane & 16) ? 8 : 0;
  const float* inimg = in + (long)img * HW;
  FragU a, bfr;
#pragma unroll
  for (int e = 0; e < 16; ++e) {
    int K = ((e < 8) ? e : (e + 8)) + off;
    float v = 0.f;
    if (K < 9) {
      int hh = h + K / 3 - 1, ww = wl + K % 3 - 1;
      if (((unsigned)hh < 64u) & ((unsigned)ww < 64u)) v = inimg[hh * WN + ww];
    }
    a.v[e] = (__bf16)v;
  }
  bfr.h[0] = *(const v8bf*)(bp + lane * 16);
  bfr.h[1] = *(const v8bf*)(bp + lane * 16 + 8);
  v8f acc = {};
  acc = __builtin_amdgcn_wmma_f32_16x16x32_bf16(false, a.v, false, bfr.v, (short)0, acc,
                                                false, false);
#pragma unroll
  for (int r = 0; r < 8; ++r) {
    int mm = mbase + r + ((lane & 16) ? 8 : 0);
    int nn = lane & 15;
    float v = acc[r] * (bng[nn] * BN_INV) + bnb[nn];
    out[((long)img * HW + mm) * 16 + nn] = (__bf16)v;
  }
}

// ---------------- conv2d 16->Cout, NHWC bf16, WMMA, LDS B staging ----------------
// 8 waves/block; each wave one 16-pixel M-tile; B (5 k-blocks) staged in LDS once.
__global__ __launch_bounds__(256)
void conv2d_nhwc_kernel(const __bf16* __restrict__ in, const __bf16* __restrict__ bp,
                        const float* __restrict__ bng, const float* __restrict__ bnb,
                        const __bf16* __restrict__ resid, __bf16* __restrict__ out,
                        int Cout, int act) {
  __shared__ __bf16 lb[5 * 512];
  int tid  = threadIdx.x;
  int lane = tid & 31;
  int wv   = tid >> 5;
  int img  = blockIdx.y;
  // stage packed B: 1280 dwords, 5 per thread
  {
    const uint32_t* s = (const uint32_t*)bp;
    uint32_t* d = (uint32_t*)lb;
#pragma unroll
    for (int i = 0; i < 5; ++i) d[tid + i * 256] = s[tid + i * 256];
  }
  __syncthreads();
  int mbase = (blockIdx.x * 8 + wv) * 16;
  int h = mbase >> 6, w0 = mbase & 63;
  int wl = w0 + (lane & 15);
  int off = (lane & 16) ? 8 : 0;
  const __bf16* inimg = in + (long)img * HW * 16;
  v8f acc = {};
#pragma unroll
  for (int cs = 0; cs < 5; ++cs) {
    FragU a, bfr;
    v8bf z = zero8();
    // low chunk: tap 2*cs, channels off..off+7 ; high chunk: tap 2*cs+1
#pragma unroll
    for (int half = 0; half < 2; ++half) {
      int tap = 2 * cs + half;
      v8bf c = z;
      if (tap < 9) {
        int hh = h + tap / 3 - 1, ww = wl + tap % 3 - 1;
        bool lv = ((unsigned)hh < 64u) & ((unsigned)ww < 64u);
        c = lv ? *(const v8bf*)(inimg + ((long)(hh * WN + ww)) * 16 + off) : z;
      }
      a.h[half] = c;
    }
    const __bf16* lp = lb + cs * 512 + lane * 16;
    bfr.h[0] = *(const v8bf*)(lp);
    bfr.h[1] = *(const v8bf*)(lp + 8);
    acc = __builtin_amdgcn_wmma_f32_16x16x32_bf16(false, a.v, false, bfr.v, (short)0, acc,
                                                  false, false);
  }
#pragma unroll
  for (int r = 0; r < 8; ++r) {
    int mm = mbase + r + ((lane & 16) ? 8 : 0);
    int nn = lane & 15;
    if (nn < Cout) {
      float v = acc[r];
      if (bng) v = v * (bng[nn] * BN_INV) + bnb[nn];
      if (act) v = (v >= 0.f) ? v : 0.1f * v;
      long o = ((long)img * HW + mm) * Cout + nn;
      if (resid) v += (float)resid[o];
      out[o] = (__bf16)v;
    }
  }
}

// ---------------- cost volume ----------------
// featV bf16 NHWC [img][4096][8]; costT bf16 [((b*HW+p)*81+a)*9+dd][8]
__global__ void build_cost_kernel(const __bf16* __restrict__ featV, const float* __restrict__ mask,
                                  __bf16* __restrict__ costT) {
  long i = (long)blockIdx.x * 256 + threadIdx.x;
  if (i >= (long)BB * HW * AA * DDN) return;
  int dd = (int)(i % 9);
  long r = i / 9;
  int a = (int)(r % 81);
  long pix = r / 81;
  int b = (int)(pix >> 12);
  int p = (int)(pix & 4095);
  int h = p >> 6, w = p & 63;
  int du = a / 9 - 4, dv = a % 9 - 4;
  int dval = dd - 4;
  int hh = h - dval * du, ww = w - dval * dv;
  float mk = mask[(long)(b * AA + a) * HW + p];
  bool ok = ((unsigned)hh < 64u) & ((unsigned)ww < 64u);
  const __bf16* f = featV + ((long)(b * AA + a) * HW + (ok ? (hh * WN + ww) : 0)) * 8;
  __bf16* o = costT + i * 8;
#pragma unroll
  for (int c = 0; c < 8; ++c) {
    float v = ok ? (float)f[c] * mk : 0.f;
    o[c] = (__bf16)v;
  }
}

// ---------------- fused per-pixel transformer ----------------
// psvG bf16 [m=(b*9+d)*HW+p][KSQR], cols 0..647 = a*8+j
__global__ void transformer_kernel(const __bf16* __restrict__ costT, const __bf16* __restrict__ featV,
                                   const float* __restrict__ wq, const float* __restrict__ wk,
                                   const float* __restrict__ wff,
                                   const float* __restrict__ ln1g, const float* __restrict__ ln1b,
                                   const float* __restrict__ ln2g, const float* __restrict__ ln2b,
                                   __bf16* __restrict__ psvG) {
  int i = blockIdx.x * 256 + threadIdx.x;
  if (i >= BB * HW * AA) return;
  int a = i % 81;
  int pix = i / 81;
  int b = pix >> 12, p = pix & 4095;
  const __bf16* fc = featV + ((long)(b * AA + 40) * HW + p) * 8;
  float ctr[8], kv[8];
#pragma unroll
  for (int c = 0; c < 8; ++c) ctr[c] = (float)fc[c];
#pragma unroll
  for (int j = 0; j < 8; ++j) {
    float s = 0.f;
    for (int c = 0; c < 8; ++c) s += wk[j * 8 + c] * ctr[c];
    kv[j] = s;
  }
  const __bf16* q = costT + (long)(pix * 81 + a) * 9 * 8;
  float lg[9];
  float mx = -1e30f;
  for (int d = 0; d < 9; ++d) {
    float s = 0.f;
    for (int j = 0; j < 8; ++j) {
      float qd = 0.f;
      for (int c = 0; c < 8; ++c) qd += wq[j * 8 + c] * (float)q[d * 8 + c];
      s += qd * kv[j];
    }
    s *= 0.35355339059327373f;
    lg[d] = s;
    mx = fmaxf(mx, s);
  }
  float sum = 0.f;
  for (int d = 0; d < 9; ++d) { lg[d] = expf(lg[d] - mx); sum += lg[d]; }
  float inv = 1.f / sum;
  for (int d = 0; d < 9; ++d) {
    float at = lg[d] * inv;
    float o[8]; float m = 0.f;
    for (int c = 0; c < 8; ++c) { o[c] = (float)q[d * 8 + c] * at; m += o[c]; }
    m *= 0.125f;
    float var = 0.f;
    for (int c = 0; c < 8; ++c) { float t = o[c] - m; var += t * t; }
    var *= 0.125f;
    float rs = rsqrtf(var + EPSF);
    float on[8];
    for (int c = 0; c < 8; ++c) on[c] = (o[c] - m) * rs * ln1g[c] + ln1b[c];
    float f2[8]; float m2 = 0.f;
    for (int j = 0; j < 8; ++j) {
      float s = 0.f;
      for (int c = 0; c < 8; ++c) s += wff[j * 8 + c] * on[c];
      f2[j] = s; m2 += s;
    }
    m2 *= 0.125f;
    float v2 = 0.f;
    for (int j = 0; j < 8; ++j) { float t = f2[j] - m2; v2 += t * t; }
    v2 *= 0.125f;
    float rs2 = rsqrtf(v2 + EPSF);
    __bf16* oo = psvG + ((long)(b * 9 + d) * HW + p) * (long)KSQR + a * 8;
    for (int j = 0; j < 8; ++j) {
      float t = (f2[j] - m2) * rs2 * ln2g[j] + ln2b[j];
      t = fmaxf(t, 0.f);
      oo[j] = (__bf16)t;
    }
  }
}

// ---------------- squeeze GEMM (648 -> 160), LDS B staging ----------------
// A bf16 [73728][KSQR]; bp fragment-packed (NT=10, KB=21); out bf16 NDHWC [m][160]
__global__ __launch_bounds__(256)
void gemm_sq_wmma_kernel(const __bf16* __restrict__ Amat, const __bf16* __restrict__ bp,
                         const float* __restrict__ bng, const float* __restrict__ bnb,
                         __bf16* __restrict__ out) {
  __shared__ __bf16 lb[21 * 512];
  int tid  = threadIdx.x;
  int lane = tid & 31;
  int wv   = tid >> 5;
  // stage this ntile's packed B: 21*512 bf16 = 5376 dwords, 21 per thread
  {
    const uint32_t* s = (const uint32_t*)(bp + (long)blockIdx.y * (21 * 512));
    uint32_t* d = (uint32_t*)lb;
#pragma unroll
    for (int i = 0; i < 21; ++i) d[tid + i * 256] = s[tid + i * 256];
  }
  __syncthreads();
  int mbase = (blockIdx.x * 8 + wv) * 16;
  int ntile = blockIdx.y * 16;
  const __bf16* arow = Amat + (long)(mbase + (lane & 15)) * KSQR;
  int off = (lane & 16) ? 8 : 0;
  v8f acc = {};
#pragma unroll
  for (int kbi = 0; kbi < 21; ++kbi) {
    FragU a, bfr;
    v8bf z = zero8();
    int s0 = kbi * 32 + off;
    int s1 = s0 + 16;
    a.h[0] = (s0 + 8 <= KSQ) ? *(const v8bf*)(arow + s0) : z;
    a.h[1] = (s1 + 8 <= KSQ) ? *(const v8bf*)(arow + s1) : z;
    const __bf16* lp = lb + kbi * 512 + lane * 16;
    bfr.h[0] = *(const v8bf*)(lp);
    bfr.h[1] = *(const v8bf*)(lp + 8);
    acc = __builtin_amdgcn_wmma_f32_16x16x32_bf16(false, a.v, false, bfr.v, (short)0, acc,
                                                  false, false);
  }
#pragma unroll
  for (int r = 0; r < 8; ++r) {
    int mm = mbase + r + ((lane & 16) ? 8 : 0);
    int nn = ntile + (lane & 15);
    float v = acc[r] * (bng[nn] * BN_INV) + bnb[nn];
    v = (v >= 0.f) ? v : 0.1f * v;
    out[(long)mm * AGGC + nn] = (__bf16)v;
  }
}

// ---------------- conv3d 3x3x3, Cin=160, NDHWC bf16, WMMA + LDS B staging ----------------
// in bf16 [m=(b*9+d)*HW+p][160]; bp fragment-packed (NT, KB=135 = 27 taps x 5); out [m][Cout]
__global__ __launch_bounds__(256)
void conv3d_nhwc_kernel(const __bf16* __restrict__ in, const __bf16* __restrict__ bp,
                        const float* __restrict__ bng, const float* __restrict__ bnb,
                        __bf16* __restrict__ out, int Cout, int act) {
  __shared__ __bf16 lb[5 * 512];
  int tid  = threadIdx.x;
  int lane = tid & 31;
  int wv   = tid >> 5;
  int mbase = (blockIdx.x * 8 + wv) * 16;
  int ntile = blockIdx.y * 16;
  int b  = mbase / (DDN * HW);
  int r2 = mbase - b * (DDN * HW);
  int d  = r2 >> 12;
  int p  = r2 & 4095;
  int h  = p >> 6, w0 = p & 63;
  int wl = w0 + (lane & 15);
  int off = (lane & 16) ? 8 : 0;
  const __bf16* bsrc = bp + (long)blockIdx.y * (135 * 512);
  v8f acc = {};
#pragma unroll 1
  for (int tap = 0; tap < 27; ++tap) {
    __syncthreads();
    {
      const uint32_t* s = (const uint32_t*)(bsrc + tap * (5 * 512));
      uint32_t* dl = (uint32_t*)lb;
#pragma unroll
      for (int i = 0; i < 5; ++i) dl[tid + i * 256] = s[tid + i * 256];
    }
    if (tap + 1 < 27)
      __builtin_prefetch(bsrc + (tap + 1) * (5 * 512) + tid * 10, 0, 1);
    __syncthreads();
    int kd = tap / 9;
    int r3 = tap - kd * 9;
    int kh = r3 / 3, kw = r3 - kh * 3;
    int dd = d + kd - 1, hh = h + kh - 1, ww = wl + kw - 1;
    bool lv = ((unsigned)dd < 9u) & ((unsigned)hh < 64u) & ((unsigned)ww < 64u);
    long abase = ((long)((b * 9 + dd) * HW + hh * WN + ww)) * AGGC;
    v8bf z = zero8();
#pragma unroll
    for (int cs = 0; cs < 5; ++cs) {
      FragU a, bfr;
      a.h[0] = lv ? *(const v8bf*)(in + abase + cs * 32 + off) : z;
      a.h[1] = lv ? *(const v8bf*)(in + abase + cs * 32 + 16 + off) : z;
      const __bf16* lp = lb + cs * 512 + lane * 16;
      bfr.h[0] = *(const v8bf*)(lp);
      bfr.h[1] = *(const v8bf*)(lp + 8);
      acc = __builtin_amdgcn_wmma_f32_16x16x32_bf16(false, a.v, false, bfr.v, (short)0, acc,
                                                    false, false);
    }
  }
#pragma unroll
  for (int r = 0; r < 8; ++r) {
    int mm = mbase + r + ((lane & 16) ? 8 : 0);
    int nn = ntile + (lane & 15);
    if (nn < Cout) {
      float v = acc[r];
      if (bng) v = v * (bng[nn] * BN_INV) + bnb[nn];
      if (act) v = (v >= 0.f) ? v : 0.1f * v;
      out[(long)mm * Cout + nn] = (__bf16)v;
    }
  }
}

// ---------------- channel attention ----------------
__global__ void ca_pool_kernel(const __bf16* __restrict__ t, float* __restrict__ y) {
  int bc = blockIdx.x; // b*160+c
  int b = bc / AGGC, c = bc - b * AGGC;
  const __bf16* src = t + (long)b * (DDN * HW) * AGGC + c;
  float s = 0.f;
  for (int i = threadIdx.x; i < DDN * HW; i += 256) s += (float)src[(long)i * AGGC];
  __shared__ float sm[256];
  sm[threadIdx.x] = s;
  __syncthreads();
  for (int o = 128; o > 0; o >>= 1) {
    if (threadIdx.x < o) sm[threadIdx.x] += sm[threadIdx.x + o];
    __syncthreads();
  }
  if (threadIdx.x == 0) y[bc] = sm[0] * (1.f / (float)(DDN * HW));
}

__global__ void ca_mlp_kernel(const float* __restrict__ y,
                              const float* __restrict__ w1, const float* __restrict__ b1,
                              const float* __restrict__ bn1g, const float* __restrict__ bn1b,
                              const float* __restrict__ w2, const float* __restrict__ b2,
                              const float* __restrict__ bn2g, const float* __restrict__ bn2b,
                              float* __restrict__ yout) {
  int b = blockIdx.x;
  int tid = threadIdx.x;
  __shared__ float y1[16];
  if (tid < 10) {
    float s = 0.f;
    for (int c = 0; c < AGGC; ++c) s += w1[tid * AGGC + c] * y[b * AGGC + c];
    s += b1[tid];
    s = s * (bn1g[tid] * BN_INV) + bn1b[tid];
    s = (s >= 0.f) ? s : 0.1f * s;
    y1[tid] = s;
  }
  __syncthreads();
  if (tid < AGGC) {
    float s = 0.f;
    for (int r = 0; r < 10; ++r) s += w2[tid * 10 + r] * y1[r];
    s += b2[tid];
    s = s * (bn2g[tid] * BN_INV) + bn2b[tid];
    yout[b * AGGC + tid] = 1.f / (1.f + expf(-s));
  }
}

// NDHWC: i = m*160+c
__global__ void ca_apply_kernel(const __bf16* __restrict__ t, const float* __restrict__ y,
                                const __bf16* __restrict__ res, __bf16* __restrict__ out) {
  long i = (long)blockIdx.x * 256 + threadIdx.x;
  if (i >= (long)BB * AGGC * DDN * HW) return;
  int c = (int)(i % AGGC);
  int b = (int)(i / ((long)AGGC * DDN * HW));
  float v = (float)t[i] * y[b * AGGC + c] + (float)res[i];
  out[i] = (__bf16)v;
}

// ---------------- final softmax-expectation ----------------
__global__ void disparity_kernel(const __bf16* __restrict__ score, float* __restrict__ out) {
  int i = blockIdx.x * 256 + threadIdx.x;
  if (i >= BB * HW) return;
  int b = i >> 12, p = i & 4095;
  float s[9], mx = -1e30f;
  for (int d = 0; d < 9; ++d) {
    s[d] = (float)score[(long)(b * 9 + d) * HW + p];
    mx = fmaxf(mx, s[d]);
  }
  float sum = 0.f;
  for (int d = 0; d < 9; ++d) { s[d] = expf(s[d] - mx); sum += s[d]; }
  float inv = 1.f / sum, r = 0.f;
  for (int d = 0; d < 9; ++d) r += s[d] * inv * (float)(d - 4);
  out[i] = r;
}

// ===================== host launcher =====================
extern "C" void kernel_launch(void* const* d_in, const int* in_sizes, int n_in,
                              void* d_out, int out_size, void* d_ws, size_t ws_size,
                              hipStream_t stream) {
  (void)in_sizes; (void)n_in; (void)out_size; (void)ws_size;
  const float* lf   = (const float*)d_in[0];
  const float* disp = (const float*)d_in[1];

  // ---- parameter leaves, jax tree-flatten order (dict keys sorted) ----
  int gi = 2;
  auto nxt = [&]() { return (const float*)d_in[gi++]; };
  const float* c1_g = nxt(); const float* c1_b = nxt(); const float* c1_w = nxt();
  const float* c2_g = nxt(); const float* c2_b = nxt(); const float* c2_w = nxt();
  const float* c3_g = nxt(); const float* c3_b = nxt(); const float* c3_w = nxt();
  const float* c4_w = nxt();
  const float* r1_bn1g = nxt(); const float* r1_bn1b = nxt();
  const float* r1_bn2g = nxt(); const float* r1_bn2b = nxt();
  const float* r1_cab1 = nxt(); const float* r1_cab2 = nxt();
  const float* r1_cabn1g = nxt(); const float* r1_cabn1b = nxt();
  const float* r1_cabn2g = nxt(); const float* r1_cabn2b = nxt();
  const float* r1_caw1 = nxt(); const float* r1_caw2 = nxt();
  const float* r1_w1 = nxt(); const float* r1_w2 = nxt();
  const float* r2_bn1g = nxt(); const float* r2_bn1b = nxt();
  const float* r2_bn2g = nxt(); const float* r2_bn2b = nxt();
  const float* r2_cab1 = nxt(); const float* r2_cab2 = nxt();
  const float* r2_cabn1g = nxt(); const float* r2_cabn1b = nxt();
  const float* r2_cabn2g = nxt(); const float* r2_cabn2b = nxt();
  const float* r2_caw1 = nxt(); const float* r2_caw2 = nxt();
  const float* r2_w1 = nxt(); const float* r2_w2 = nxt();
  const float* sq_g = nxt(); const float* sq_b = nxt(); const float* sq_w = nxt();
  const float* ln1g = nxt(); const float* ln1b = nxt();
  const float* ln2g = nxt(); const float* ln2b = nxt();
  const float* wff = nxt(); const float* wk = nxt(); const float* wq = nxt();
  const float* bn0g = nxt(); const float* bn0b = nxt();
  const float* bn1g = nxt(); const float* bn1b = nxt();
  const float *rb_g1[8], *rb_b1[8], *rb_g2[8], *rb_b2[8], *rb_w1[8], *rb_w2[8];
  for (int j = 0; j < 8; ++j) {
    rb_g1[j] = nxt(); rb_b1[j] = nxt();
    rb_g2[j] = nxt(); rb_b2[j] = nxt();
    rb_w1[j] = nxt(); rb_w2[j] = nxt();
  }
  const float* w0  = nxt();
  const float* w1i = nxt();
  const float* w2i = nxt();

  // ---- workspace bump allocator ----
  uintptr_t base = (uintptr_t)d_ws;
  size_t off = 0;
  auto alloc = [&](size_t bytes) -> void* {
    off = (off + 255) & ~(size_t)255;
    void* p = (void*)(base + off);
    off += bytes;
    return p;
  };

  float*  maskW = (float*)alloc((size_t)BB * AA * HW * 4);                 // 2.65 MB
  __bf16* featV = (__bf16*)alloc((size_t)NIMG * HW * FEATC * 2);           // 10.6 MB

  // fragment-packed weights (every element written by pack kernels)
  __bf16* bp_l0 = (__bf16*)alloc((size_t)512 * 2);
  __bf16* bp_rb = (__bf16*)alloc((size_t)16 * 2560 * 2);
  __bf16* bp_w1 = (__bf16*)alloc((size_t)2560 * 2);
  __bf16* bp_w2 = (__bf16*)alloc((size_t)2560 * 2);
  __bf16* bp_sq = (__bf16*)alloc((size_t)10 * 21 * 512 * 2);               // 107520
  __bf16* bp_c[7];
  for (int j = 0; j < 7; ++j) bp_c[j] = (__bf16*)alloc((size_t)10 * 135 * 512 * 2); // 691200 each
  __bf16* bp_c4 = (__bf16*)alloc((size_t)1 * 135 * 512 * 2);               // 69120

  float*  ypool  = (float*)alloc((size_t)BB * AGGC * 4);
  float*  yatt   = (float*)alloc((size_t)BB * AGGC * 4);
  __bf16* scoreB = (__bf16*)alloc((size_t)M3D * 2);

  const size_t tmpElems = (size_t)NIMG * HW * 16;        // NHWC bf16 feature tmp
  const size_t costElems = (size_t)BB * HW * AA * DDN * 8;
  const size_t aggElems = (size_t)M3D * AGGC;
  size_t arenaABytes = costElems * 2;                    // 95.6 MB (largest occupant)
  if (arenaABytes < 3 * tmpElems * 2) arenaABytes = 3 * tmpElems * 2;
  if (arenaABytes < 3 * aggElems * 2) arenaABytes = 3 * aggElems * 2;
  char* arenaA = (char*)alloc(arenaABytes);
  char* arenaB = (char*)alloc((size_t)M3D * KSQR * 2);   // psvG, 96.7 MB
  __bf16* tmpA = (__bf16*)arenaA;
  __bf16* tmpB = tmpA + tmpElems;
  __bf16* tmpC = tmpB + tmpElems;
  __bf16* costT = (__bf16*)arenaA;                       // after feature tmps die
  __bf16* psvG  = (__bf16*)arenaB;
  __bf16* xa = (__bf16*)arenaA;                          // after costT dies
  __bf16* xb = xa + aggElems;
  __bf16* xc = xb + aggElems;

  auto cdiv = [](long n, long d) { return (int)((n + d - 1) / d); };

  // ---- weight packing ----
  pack_b_frag_kernel<<<2, 256, 0, stream>>>(w0, bp_l0, 16, 1, 9, 9, 1, 512);
  for (int j = 0; j < 8; ++j) {
    pack_b_frag_kernel<<<10, 256, 0, stream>>>(rb_w1[j], bp_rb + (size_t)(2 * j) * 2560, 16, 16, 9, K2D, 5, 2560);
    pack_b_frag_kernel<<<10, 256, 0, stream>>>(rb_w2[j], bp_rb + (size_t)(2 * j + 1) * 2560, 16, 16, 9, K2D, 5, 2560);
  }
  pack_b_frag_kernel<<<10, 256, 0, stream>>>(w1i, bp_w1, 16, 16, 9, K2D, 5, 2560);
  pack_b_frag_kernel<<<10, 256, 0, stream>>>(w2i, bp_w2, 8, 16, 9, K2D, 5, 2560);
  pack_b_frag_kernel<<<cdiv(107520, 256), 256, 0, stream>>>(sq_w, bp_sq, AGGC, KSQ, 1, KSQ, 21, 107520);
  const float* cw[7] = {c1_w, c2_w, c3_w, r1_w1, r1_w2, r2_w1, r2_w2};
  for (int j = 0; j < 7; ++j)
    pack_b_frag_kernel<<<cdiv(691200, 256), 256, 0, stream>>>(cw[j], bp_c[j], AGGC, AGGC, 27, K3D, 135, 691200);
  pack_b_frag_kernel<<<cdiv(69120, 256), 256, 0, stream>>>(c4_w, bp_c4, 1, AGGC, 27, K3D, 135, 69120);

  // ---- mask ----
  mask_kernel<<<cdiv((long)BB * AA * HW, 256), 256, 0, stream>>>(lf, disp, maskW);

  // ---- init_feature ----
  conv2d_c1_kernel<<<dim3(256, NIMG), 32, 0, stream>>>(lf, bp_l0, bn0g, bn0b, tmpA);
  dim3 g2(32, NIMG);
  __bf16* curY = tmpA;
  __bf16* freeT = tmpC;
  for (int j = 0; j < 8; ++j) {
    conv2d_nhwc_kernel<<<g2, 256, 0, stream>>>(curY, bp_rb + (size_t)(2 * j) * 2560,
                                               rb_g1[j], rb_b1[j], nullptr, tmpB, 16, 1);
    conv2d_nhwc_kernel<<<g2, 256, 0, stream>>>(tmpB, bp_rb + (size_t)(2 * j + 1) * 2560,
                                               rb_g2[j], rb_b2[j], curY, freeT, 16, 0);
    __bf16* t = curY; curY = freeT; freeT = t;
  }
  conv2d_nhwc_kernel<<<g2, 256, 0, stream>>>(curY, bp_w1, bn1g, bn1b, nullptr, tmpB, 16, 1);
  conv2d_nhwc_kernel<<<g2, 256, 0, stream>>>(tmpB, bp_w2, nullptr, nullptr, nullptr, featV, 8, 0);

  // ---- cost volume (feature tmps dead -> costT) ----
  build_cost_kernel<<<cdiv((long)BB * HW * AA * DDN, 256), 256, 0, stream>>>(featV, maskW, costT);

  // ---- transformer -> psv ----
  transformer_kernel<<<cdiv((long)BB * HW * AA, 256), 256, 0, stream>>>(
      costT, featV, wq, wk, wff, ln1g, ln1b, ln2g, ln2b, psvG);

  // ---- aggregate ----
  dim3 g3(M3D / 128, AGGC / 16);
  gemm_sq_wmma_kernel<<<g3, 256, 0, stream>>>(psvG, bp_sq, sq_g, sq_b, xa);

  conv3d_nhwc_kernel<<<g3, 256, 0, stream>>>(xa, bp_c[0], c1_g, c1_b, xb, AGGC, 1);
  conv3d_nhwc_kernel<<<g3, 256, 0, stream>>>(xb, bp_c[1], c2_g, c2_b, xc, AGGC, 1);
  // r1
  conv3d_nhwc_kernel<<<g3, 256, 0, stream>>>(xc, bp_c[3], r1_bn1g, r1_bn1b, xa, AGGC, 1);
  conv3d_nhwc_kernel<<<g3, 256, 0, stream>>>(xa, bp_c[4], r1_bn2g, r1_bn2b, xb, AGGC, 0);
  ca_pool_kernel<<<BB * AGGC, 256, 0, stream>>>(xb, ypool);
  ca_mlp_kernel<<<BB, AGGC, 0, stream>>>(ypool, r1_caw1, r1_cab1, r1_cabn1g, r1_cabn1b,
                                         r1_caw2, r1_cab2, r1_cabn2g, r1_cabn2b, yatt);
  ca_apply_kernel<<<cdiv((long)BB * AGGC * DDN * HW, 256), 256, 0, stream>>>(xb, yatt, xc, xa);
  // r2
  conv3d_nhwc_kernel<<<g3, 256, 0, stream>>>(xa, bp_c[5], r2_bn1g, r2_bn1b, xb, AGGC, 1);
  conv3d_nhwc_kernel<<<g3, 256, 0, stream>>>(xb, bp_c[6], r2_bn2g, r2_bn2b, xc, AGGC, 0);
  ca_pool_kernel<<<BB * AGGC, 256, 0, stream>>>(xc, ypool);
  ca_mlp_kernel<<<BB, AGGC, 0, stream>>>(ypool, r2_caw1, r2_cab1, r2_cabn1g, r2_cabn1b,
                                         r2_caw2, r2_cab2, r2_cabn2g, r2_cabn2b, yatt);
  ca_apply_kernel<<<cdiv((long)BB * AGGC * DDN * HW, 256), 256, 0, stream>>>(xc, yatt, xa, xb);
  // c3, c4
  conv3d_nhwc_kernel<<<g3, 256, 0, stream>>>(xb, bp_c[2], c3_g, c3_b, xa, AGGC, 1);
  conv3d_nhwc_kernel<<<dim3(M3D / 128, 1), 256, 0, stream>>>(xa, bp_c4, nullptr, nullptr, scoreB, 1, 0);

  // ---- disparity regression ----
  disparity_kernel<<<cdiv((long)BB * HW, 256), 256, 0, stream>>>(scoreB, (float*)d_out);
}
